// SoftNeighSuperpixelAgg_85117661872429
// MI455X (gfx1250) — matrix operand order repaired and checked by reading, not compiled, not run on hardware
//
#include <hip/hip_runtime.h>

// ---------------------------------------------------------------------------
// SSNA forward for MI455X (gfx1250, wave32, WMMA bf16 16x16x32)
//   K0: cvt x, v_w, proj_w -> bf16 (hoists ALL f32->bf16 out of GEMM loops)
//   K1: v   = xbf @ vwbf^T + v_b            (pure load+WMMA, out bf16 -> d_out scratch)
//   K2: wgt[b,h,w,49]                        (superpixel soft-match, head-independent)
//   K3: out_agg = sum_k attn*wgt * v_nbr     (LDS halo tile, bf16 -> ws, overwrites xbf)
//   K4: out = aggbf @ pwbf^T + proj_b        (pure load+WMMA, fp32 -> d_out)
// ---------------------------------------------------------------------------

typedef __attribute__((ext_vector_type(16))) __bf16 v16bf;
typedef __attribute__((ext_vector_type(8)))  float  v8f;

#define Bsz  8
#define Hh   128
#define Ww   128
#define Cc   512
#define NHh  16
#define HDd  32
#define Kk   7
#define KK   49
#define Ss   9
#define Mtot (Bsz*Hh*Ww)      // 131072 pixel-rows

union Frag { unsigned u[8]; v16bf v; };

// hardware converts (v_cvt_*_bf16_f32), not bit-twiddled emulation
static __device__ __forceinline__ unsigned short f2bf(float f) {
    return __builtin_bit_cast(unsigned short, (__bf16)f);
}
static __device__ __forceinline__ unsigned packbf(float lo, float hi) {
    union { __bf16 h[2]; unsigned u; } x;
    x.h[0] = (__bf16)lo; x.h[1] = (__bf16)hi;
    return x.u;
}

// ---------------------------------------------------------------------------
// K0: elementwise f32 -> bf16, 8 elements/thread (float4 in, uint2 out)
// ---------------------------------------------------------------------------
__global__ __launch_bounds__(256)
void cvt_bf16_kernel(const float* __restrict__ in, unsigned short* __restrict__ out,
                     long long n8) {                       // n8 = n/8
    long long i = (long long)blockIdx.x * blockDim.x + threadIdx.x;
    if (i >= n8) return;
    const float4* p = (const float4*)(in) + i * 2;
    float4 f0 = p[0], f1 = p[1];
    uint2 o;
    o.x = packbf(f0.x, f0.y);
    o.y = packbf(f0.z, f0.w);
    ((uint2*)out)[i * 2 + 0] = o;
    o.x = packbf(f1.x, f1.y);
    o.y = packbf(f1.z, f1.w);
    ((uint2*)out)[i * 2 + 1] = o;
}

// ---------------------------------------------------------------------------
// GEMM + bias: Out[M,512] = A[M,512] @ W[512,512]^T + bias, all-bf16 operands.
// block = 256 thr = 8 waves; each wave: 32 rows x 64 cols (2x4 wmma tiles).
// Block covers 256 rows x 64 cols. grid = (N/64, M/256).
// Inner loop: 2 A-frag loads (2x2xb128) + 4 B-frag loads (4x2xb128), 8 WMMAs.
// ---------------------------------------------------------------------------
template<int OUT_BF16>
__global__ __launch_bounds__(256)
void wmma_gemm_bias(const unsigned short* __restrict__ A,
                    const unsigned short* __restrict__ W,
                    const float* __restrict__ bias, void* __restrict__ Outp) {
    const int lane = threadIdx.x & 31;
    const int wave = threadIdx.x >> 5;
    const int l16  = lane & 15;
    const int half = lane >> 4;

    const int row0 = blockIdx.y * 256 + wave * 32;   // wave: rows row0..row0+31
    const int col0 = blockIdx.x * 64;

    v8f c00 = {}, c01 = {}, c02 = {}, c03 = {};
    v8f c10 = {}, c11 = {}, c12 = {}, c13 = {};

    const uint4* a0p = (const uint4*)(A + (size_t)(row0 + l16) * Cc);
    const uint4* a1p = (const uint4*)(A + (size_t)(row0 + 16 + l16) * Cc);

    for (int kk = 0; kk < Cc; kk += 32) {
        // A fragment (16x32 bf16): lane-half selects K {0..15} / {16..31} (contiguous)
        const int ia0 = (kk + half * 16) >> 3;       // uint4 = 8 bf16
        const int ia1 = (kk + half * 16 + 8) >> 3;
        Frag a0, a1;
        { uint4 u0 = a0p[ia0], u1 = a0p[ia1];
          a0.u[0]=u0.x; a0.u[1]=u0.y; a0.u[2]=u0.z; a0.u[3]=u0.w;
          a0.u[4]=u1.x; a0.u[5]=u1.y; a0.u[6]=u1.z; a0.u[7]=u1.w; }
        { uint4 u0 = a1p[ia0], u1 = a1p[ia1];
          a1.u[0]=u0.x; a1.u[1]=u0.y; a1.u[2]=u0.z; a1.u[3]=u0.w;
          a1.u[4]=u1.x; a1.u[5]=u1.y; a1.u[6]=u1.z; a1.u[7]=u1.w; }

        // B fragments (32x16): lane<16 -> col N=lane, K=kbase..+15 (contiguous)
        const int kbase = kk + half * 16;
        Frag b[4];
        #pragma unroll
        for (int j = 0; j < 4; ++j) {
            const uint4* bp = (const uint4*)(W + (size_t)(col0 + j * 16 + l16) * Cc + kbase);
            uint4 u0 = bp[0], u1 = bp[1];
            b[j].u[0]=u0.x; b[j].u[1]=u0.y; b[j].u[2]=u0.z; b[j].u[3]=u0.w;
            b[j].u[4]=u1.x; b[j].u[5]=u1.y; b[j].u[6]=u1.z; b[j].u[7]=u1.w;
        }

        c00 = __builtin_amdgcn_wmma_f32_16x16x32_bf16(false, a0.v, false, b[0].v, (short)0, c00, false, false);
        c01 = __builtin_amdgcn_wmma_f32_16x16x32_bf16(false, a0.v, false, b[1].v, (short)0, c01, false, false);
        c02 = __builtin_amdgcn_wmma_f32_16x16x32_bf16(false, a0.v, false, b[2].v, (short)0, c02, false, false);
        c03 = __builtin_amdgcn_wmma_f32_16x16x32_bf16(false, a0.v, false, b[3].v, (short)0, c03, false, false);
        c10 = __builtin_amdgcn_wmma_f32_16x16x32_bf16(false, a1.v, false, b[0].v, (short)0, c10, false, false);
        c11 = __builtin_amdgcn_wmma_f32_16x16x32_bf16(false, a1.v, false, b[1].v, (short)0, c11, false, false);
        c12 = __builtin_amdgcn_wmma_f32_16x16x32_bf16(false, a1.v, false, b[2].v, (short)0, c12, false, false);
        c13 = __builtin_amdgcn_wmma_f32_16x16x32_bf16(false, a1.v, false, b[3].v, (short)0, c13, false, false);
    }

    // D layout: lane<16: VGPR r -> M=r ; lane>=16: VGPR r -> M=8+r ; N = l16
    #pragma unroll
    for (int t = 0; t < 8; ++t) {
        const int mt = t >> 2, j = t & 3;
        v8f acc;
        switch (t) {
            case 0: acc = c00; break; case 1: acc = c01; break;
            case 2: acc = c02; break; case 3: acc = c03; break;
            case 4: acc = c10; break; case 5: acc = c11; break;
            case 6: acc = c12; break; default: acc = c13; break;
        }
        const int n  = col0 + j * 16 + l16;
        const float bv = bias[n];
        #pragma unroll
        for (int r = 0; r < 8; ++r) {
            const int m = row0 + mt * 16 + half * 8 + r;
            float val = acc[r] + bv;
            if (OUT_BF16) ((unsigned short*)Outp)[(size_t)m * Cc + n] = f2bf(val);
            else          ((float*)Outp)[(size_t)m * Cc + n] = val;
        }
    }
}

// ---------------------------------------------------------------------------
// wgt[b,h,w,k] = sum_{s,t} sims_p[s]*sims_q[t]*[sinds_p[s]==sinds_q[t]]
// LDS-tiled: 22x22 halo of (9 sims + 9 sinds). grid=(64 tiles, B), block=256.
// ---------------------------------------------------------------------------
__global__ __launch_bounds__(256)
void wgt_kernel(const float* __restrict__ sims, const int* __restrict__ sinds,
                float* __restrict__ wgt) {
    __shared__ float s_sims[484 * 9];
    __shared__ int   s_inds[484 * 9];
    const int b    = blockIdx.y;
    const int tile = blockIdx.x;
    const int tr = (tile >> 3) * 16, tc = (tile & 7) * 16;
    const int rb = min(max(tr - 3, 0), Hh - 22);
    const int cb = min(max(tc - 3, 0), Ww - 22);

    for (int p = threadIdx.x; p < 484; p += 256) {
        int gr = rb + p / 22, gc = cb + p % 22;
        size_t g = ((size_t)(b * Hh + gr) * Ww + gc) * Ss;
        #pragma unroll
        for (int s = 0; s < Ss; ++s) {
            s_sims[p * 9 + s] = sims[g + s];
            s_inds[p * 9 + s] = sinds[g + s];
        }
    }
    __syncthreads();

    const int h = tr + (threadIdx.x >> 4);
    const int w = tc + (threadIdx.x & 15);
    const int si = min(max(h - 3, 0), Hh - Kk);
    const int sj = min(max(w - 3, 0), Ww - Kk);
    const int lp = ((h - rb) * 22 + (w - cb)) * 9;

    float ps[9]; int pi[9];
    #pragma unroll
    for (int s = 0; s < 9; ++s) { ps[s] = s_sims[lp + s]; pi[s] = s_inds[lp + s]; }

    float* wp = wgt + ((size_t)(b * Hh + h) * Ww + w) * KK;
    for (int k = 0; k < KK; ++k) {
        int ni = si + k / Kk, nj = sj + k % Kk;
        int lq = ((ni - rb) * 22 + (nj - cb)) * 9;
        float acc = 0.f;
        #pragma unroll
        for (int s = 0; s < 9; ++s) {
            float inner = 0.f;
            #pragma unroll
            for (int t = 0; t < 9; ++t)
                inner += (pi[s] == s_inds[lq + t]) ? s_sims[lq + t] : 0.f;
            acc += ps[s] * inner;
        }
        wp[k] = acc;
    }
}

// ---------------------------------------------------------------------------
// Aggregation: one (b, head) per block, 16x16 pixels, 22x22x32ch bf16 halo in
// LDS (31 KB of the 320 KB WGP LDS). Per-thread fp32 acc[32].
// grid = (64 tiles, B*NH), block = 256.
// ---------------------------------------------------------------------------
__global__ __launch_bounds__(256)
void agg_kernel(const unsigned short* __restrict__ vbf,   // [B,H,W,NH,HD] bf16
                const float* __restrict__ attn,           // [B,NH,H,W,49]
                const float* __restrict__ wgt,            // [B,H,W,49]
                unsigned short* __restrict__ aggbf) {     // [B,H,W,NH,HD] bf16
    __shared__ uint4 vt[484 * 4];                         // 30976 B
    const int bh = blockIdx.y;                            // b*NH + nh
    const int b = bh >> 4, nh = bh & 15;
    const int tile = blockIdx.x;
    const int tr = (tile >> 3) * 16, tc = (tile & 7) * 16;
    const int rb = min(max(tr - 3, 0), Hh - 22);
    const int cb = min(max(tc - 3, 0), Ww - 22);

    const uint4* vsrc = (const uint4*)vbf;
    for (int p = threadIdx.x; p < 484; p += 256) {
        int gr = rb + p / 22, gc = cb + p % 22;
        size_t g4 = (((size_t)(b * Hh + gr) * Ww + gc) * NHh + nh) * 4;
        vt[p * 4 + 0] = vsrc[g4 + 0];
        vt[p * 4 + 1] = vsrc[g4 + 1];
        vt[p * 4 + 2] = vsrc[g4 + 2];
        vt[p * 4 + 3] = vsrc[g4 + 3];
    }
    __syncthreads();

    const int h = tr + (threadIdx.x >> 4);
    const int w = tc + (threadIdx.x & 15);
    const int si = min(max(h - 3, 0), Hh - Kk);
    const int sj = min(max(w - 3, 0), Ww - Kk);
    const size_t pix = (size_t)(b * Hh + h) * Ww + w;

    const float* ap = attn + ((size_t)bh * Hh * Ww + (size_t)h * Ww + w) * KK;
    const float* wp = wgt + pix * KK;

    float acc[HDd];
    #pragma unroll
    for (int i = 0; i < HDd; ++i) acc[i] = 0.f;

    for (int k = 0; k < KK; ++k) {
        int ni = si + k / Kk, nj = sj + k % Kk;
        int lq = ((ni - rb) * 22 + (nj - cb)) * 4;
        float coef = ap[k] * wp[k];
        #pragma unroll
        for (int q = 0; q < 4; ++q) {
            uint4 d = vt[lq + q];
            unsigned uu[4] = { d.x, d.y, d.z, d.w };
            #pragma unroll
            for (int e = 0; e < 4; ++e) {
                unsigned u = uu[e];
                float lo = __builtin_bit_cast(float, u << 16);
                float hi = __builtin_bit_cast(float, u & 0xFFFF0000u);
                acc[q * 8 + e * 2 + 0] += coef * lo;
                acc[q * 8 + e * 2 + 1] += coef * hi;
            }
        }
    }

    uint4* o4 = (uint4*)(aggbf + (pix * NHh + nh) * HDd);
    #pragma unroll
    for (int q = 0; q < 4; ++q) {
        uint4 d;
        d.x = packbf(acc[q * 8 + 0], acc[q * 8 + 1]);
        d.y = packbf(acc[q * 8 + 2], acc[q * 8 + 3]);
        d.z = packbf(acc[q * 8 + 4], acc[q * 8 + 5]);
        d.w = packbf(acc[q * 8 + 6], acc[q * 8 + 7]);
        o4[q] = d;
    }
}

// ---------------------------------------------------------------------------
extern "C" void kernel_launch(void* const* d_in, const int* in_sizes, int n_in,
                              void* d_out, int out_size, void* d_ws, size_t ws_size,
                              hipStream_t stream) {
    const float* x     = (const float*)d_in[0];
    const float* attn  = (const float*)d_in[1];
    const float* sims  = (const float*)d_in[2];
    const int*   sinds = (const int*)d_in[3];
    const float* v_w   = (const float*)d_in[4];
    const float* v_b   = (const float*)d_in[5];
    const float* p_w   = (const float*)d_in[6];
    const float* p_b   = (const float*)d_in[7];

    // v (bf16, 128 MB) staged in d_out (256 MB fp32) -- fully overwritten by K4.
    unsigned short* vbf = (unsigned short*)d_out;

    // ws layout: [ regionA: xbf16 (K0/K1) then aggbf (K3/K4), 128 MB ]
    //            [ wgt fp32, 25.7 MB ] [ vw bf16, 512 KB ] [ pw bf16, 512 KB ]
    const size_t SZ_A = (size_t)Mtot * Cc * 2;             // 134,217,728
    const size_t SZ_W = (size_t)Mtot * KK * 4;             //  25,690,112
    unsigned short* xbf  = (unsigned short*)d_ws;
    unsigned short* aggb = (unsigned short*)d_ws;          // reuses xbf region
    float*          wgt  = (float*)((char*)d_ws + SZ_A);
    unsigned short* vwbf = (unsigned short*)((char*)d_ws + SZ_A + SZ_W);
    unsigned short* pwbf = vwbf + (size_t)Cc * Cc;

    dim3 blk(256);

    // K0: bf16 conversions (x: 64M elems; weights: 256K elems each)
    {
        long long n8 = (long long)Mtot * Cc / 8;
        cvt_bf16_kernel<<<dim3((unsigned)((n8 + 255) / 256)), blk, 0, stream>>>(x, xbf, n8);
        long long w8 = (long long)Cc * Cc / 8;
        cvt_bf16_kernel<<<dim3((unsigned)((w8 + 255) / 256)), blk, 0, stream>>>(v_w, vwbf, w8);
        cvt_bf16_kernel<<<dim3((unsigned)((w8 + 255) / 256)), blk, 0, stream>>>(p_w, pwbf, w8);
    }

    dim3 ggrid(Cc / 64, Mtot / 256);       // (8, 512)
    wmma_gemm_bias<1><<<ggrid, blk, 0, stream>>>(xbf, vwbf, v_b, vbf);

    dim3 wgrid(64, Bsz);
    wgt_kernel<<<wgrid, blk, 0, stream>>>(sims, sinds, wgt);

    dim3 agrid(64, Bsz * NHh);
    agg_kernel<<<agrid, blk, 0, stream>>>(vbf, attn, wgt, aggb);   // overwrites xbf (dead)

    wmma_gemm_bias<0><<<ggrid, blk, 0, stream>>>(aggb, pwbf, p_b, (float*)d_out);
}
